// AttModel_24670292148765
// MI455X (gfx1250) — matrix operand order repaired and checked by reading, not compiled, not run on hardware
//
#include <hip/hip_runtime.h>
#include <hip/hip_bf16.h>

typedef __attribute__((ext_vector_type(16))) __bf16 v16bf;
typedef __attribute__((ext_vector_type(8)))  float  v8f;

#define WAVES 8

// ---------------- LDS layout (bytes) ----------------
constexpr int OFF_Y     = 0;                    // f32  [54][512]  residual stream
constexpr int OFF_YBFT  = 110592;               // bf16 [512][64]  Y^T (GEMM1 B-operand)
constexpr int OFF_TBF   = 176128;               // bf16 [64][512]  T = att@X (GEMM2 A-operand)
constexpr int OFF_HBFT  = 241664;               // bf16 [512][64]  H^T
constexpr int OFF_DCT20 = 307200;               // f32  [54][32]   dct_in (concat, 20 valid cols)
constexpr int OFF_DCTM  = 314112;               // f32  [10][16]   DCT matrix (15 valid cols)
constexpr int OFF_ATT   = 314752;               // f32  [64]       scores / att weights
constexpr int OFF_QF    = 315008;               // f32  [512]      query features
constexpr int LDS_SIZE  = 317056;
// conv-phase overlays (dead before GCN starts)
constexpr int OFF_SRCP  = OFF_Y;                // bf16 [64][64]   src/1000 (time x chan, padded)
constexpr int OFF_SRCF  = OFF_Y + 8192;         // f32  [50][56]   raw src
constexpr int OFF_SBUF  = OFF_Y + 8192 + 11200; // f32  [15][56]   attention-weighted windows

// ---------------- fragment loaders ----------------
// A 16x32 bf16: lane m=l&15, kb = (l>=16)?8:0 ; elems 0..7 -> K=kb.., 8..15 -> K=kb+16..
__device__ __forceinline__ v16bf ld_fragA(const __bf16* p) {
  union { v16bf v; uint4 q[2]; } u;
  u.q[0] = *(const uint4*)(p);
  u.q[1] = *(const uint4*)(p + 16);
  return u.v;
}
// B 32x16 bf16 (as B^T rows): lane n=l&15, kb = (l>=16)?8:0 ; elems 0..7 -> K=kb.., 8..15 -> K=kb+16..
__device__ __forceinline__ v16bf ld_fragB(const __bf16* p) {
  union { v16bf v; uint4 q[2]; } u;
  u.q[0] = *(const uint4*)(p);
  u.q[1] = *(const uint4*)(p + 16);
  return u.v;
}

// C = A(MxK) * B(KxN); A row-major, B given transposed (B^T: N x K row-major).
// taps>1 implements conv-as-GEMM: A advances by tapStrideA, B^T row shifts by tap.
template <typename Epi>
__device__ __forceinline__ void run_gemm(
    const __bf16* __restrict__ A, int ldA,
    const __bf16* __restrict__ BT, int ldB,
    int Mt, int Nt, int Kc, int taps, int tapStrideA,
    int wave, int lane, Epi epi)
{
  const int m   = lane & 15;
  const int kbA = (lane & 16) ? 8 : 0;
  const int kbB = (lane & 16) ? 8 : 0;
  for (int t = wave; t < Mt * Nt; t += WAVES) {
    int mt = t / Nt, nt = t - mt * Nt;
    v8f c = {};
    for (int tap = 0; tap < taps; ++tap) {
      const __bf16* ar = A + tap * tapStrideA + (mt * 16 + m) * ldA + kbA;
      const __bf16* br = BT + (nt * 16 + m + tap) * ldB + kbB;
#pragma unroll 4
      for (int kc = 0; kc < Kc; ++kc) {
        v16bf av = ld_fragA(ar + kc * 32);
        v16bf bv = ld_fragB(br + kc * 32);
        c = __builtin_amdgcn_wmma_f32_16x16x32_bf16(false, av, false, bv,
                                                    (short)0, c, false, false);
      }
    }
    epi(mt, nt, c);
  }
}

// ---------------- weight prep kernels (f32 -> padded bf16, run on L2-resident data) ------
__global__ void prep_conv_w1(const float* __restrict__ w, __bf16* __restrict__ dst) {
  int i = blockIdx.x * 256 + threadIdx.x;             // [3][512][64]
  if (i >= 3 * 512 * 64) return;
  int tap = i / (512 * 64);
  int rem = i - tap * 512 * 64;
  int o = rem >> 6, c = rem & 63;
  float v = (c < 54) ? w[(o * 54 + c) * 3 + tap] : 0.f;
  dst[i] = (__bf16)v;
}
__global__ void prep_conv_w2(const float* __restrict__ w, __bf16* __restrict__ dst) {
  int i = blockIdx.x * 256 + threadIdx.x;             // [3][512][512]
  if (i >= 3 * 512 * 512) return;
  int tap = i / (512 * 512);
  int rem = i - tap * 512 * 512;
  int o = rem >> 9, c = rem & 511;
  dst[i] = (__bf16)w[(o * 512 + c) * 3 + tap];
}
__global__ void prep_att64(const float* __restrict__ a, __bf16* __restrict__ dst) {
  int i = blockIdx.x * 256 + threadIdx.x;             // [64][64]
  if (i >= 4096) return;
  int r = i >> 6, c = i & 63;
  dst[i] = (__bf16)((r < 54 && c < 54) ? a[r * 54 + c] : 0.f);
}
// w is (I x O) row-major; dst = W^T padded to [Rp][Cp]: dst[o][i]
__global__ void prep_wT(const float* __restrict__ w, __bf16* __restrict__ dst,
                        int I, int O, int Rp, int Cp) {
  int i = blockIdx.x * 256 + threadIdx.x;
  if (i >= Rp * Cp) return;
  int r = i / Cp, c = i - r * Cp;
  dst[i] = (__bf16)((r < O && c < I) ? w[c * O + r] : 0.f);
}

// ---------------- kernel params ----------------
struct KParams {
  const float* src;
  float* out;
  const __bf16 *WK1, *WQ1, *WK2, *WQ2;
  const __bf16 *ATT0, *ATT7;
  const __bf16 *ATTS[4];
  const __bf16 *W1T, *W7T;
  const __bf16 *WB[4];
  const float *B1, *G1, *BB1;
  const float *SB[4], *SG[4], *SBB[4];
  const float *B7;
};

#define BN_INV 0.9999950000374997f

__global__ __launch_bounds__(256) void attmodel_main(KParams P) {
  extern __shared__ __attribute__((aligned(16))) char smem[];
  const int tid  = threadIdx.x;
  const int lane = tid & 31;
  const int wave = tid >> 5;
  const int bi   = blockIdx.x;

  float*  Yf    = (float*)(smem + OFF_Y);
  __bf16* YbfT  = (__bf16*)(smem + OFF_YBFT);
  __bf16* Tbf   = (__bf16*)(smem + OFF_TBF);
  __bf16* HbfT  = (__bf16*)(smem + OFF_HBFT);
  float*  dct20 = (float*)(smem + OFF_DCT20);
  float*  DCTm  = (float*)(smem + OFF_DCTM);
  float*  attw  = (float*)(smem + OFF_ATT);
  float*  qf    = (float*)(smem + OFF_QF);
  __bf16* srcP  = (__bf16*)(smem + OFF_SRCP);
  float*  srcF  = (float*)(smem + OFF_SRCF);
  float*  sbuf  = (float*)(smem + OFF_SBUF);
  __bf16* a1t   = Tbf;                     // conv1 output overlay (time x 512)
  __bf16* kft   = YbfT;                    // keyf overlay (time x 512)
  __bf16* x0t   = HbfT;                    // dct_in^T overlay ([32][64])
  float*  out20 = (float*)(smem + OFF_HBFT); // gc7 output overlay ([64][32] f32)

  // zero all LDS once: guarantees all zero-padding regions are finite zeros
  for (int i = tid; i < LDS_SIZE / 4; i += 256) ((unsigned*)smem)[i] = 0u;
  __syncthreads();

  // orthonormal DCT-II (inverse = transpose)
  for (int i = tid; i < 160; i += 256) {
    int d = i >> 4, l = i & 15;
    float v = 0.f;
    if (l < 15) {
      float w = (d == 0) ? 0.25819888974716115f : 0.36514837167011072f;
      v = w * cosf(3.14159265358979323846f * (l + 0.5f) * (float)d / 15.0f);
    }
    DCTm[i] = v;
  }
  const float* srcb = P.src + (size_t)bi * 2700;  // (50,54)
  for (int i = tid; i < 2700; i += 256) {
    int r = i / 54, c = i - r * 54;
    float v = srcb[i];
    srcF[r * 56 + c] = v;
    srcP[r * 64 + c] = (__bf16)(v * 0.001f);
  }
  __syncthreads();

  // ---- conv stacks (conv1d as tap-accumulated GEMM) ----
  // convK1: (54->512, len 40->38)  M=512, N=48(pad of 38), K=64(pad of 54), 3 taps
  run_gemm(P.WK1, 64, srcP, 64, 32, 3, 2, 3, 512 * 64, wave, lane,
    [&](int mt, int nt, v8f c) {
      int gn = nt * 16 + (lane & 15), half = (lane >> 4) & 1;
#pragma unroll
      for (int r = 0; r < 8; ++r) {
        int gm = mt * 16 + r + half * 8;
        a1t[gn * 512 + gm] = (__bf16)fmaxf(c[r], 0.f);
      }
    });
  __syncthreads();
  // convK2: (512->512, len 38->36) K=512 x 3 taps
  run_gemm(P.WK2, 512, a1t, 512, 32, 3, 16, 3, 512 * 512, wave, lane,
    [&](int mt, int nt, v8f c) {
      int gn = nt * 16 + (lane & 15), half = (lane >> 4) & 1;
#pragma unroll
      for (int r = 0; r < 8; ++r) {
        int gm = mt * 16 + r + half * 8;
        kft[gn * 512 + gm] = (__bf16)fmaxf(c[r], 0.f);
      }
    });
  __syncthreads();
  // convQ1: times 45..49 (len 5->3)
  run_gemm(P.WQ1, 64, srcP + 45 * 64, 64, 32, 1, 2, 3, 512 * 64, wave, lane,
    [&](int mt, int nt, v8f c) {
      int gn = nt * 16 + (lane & 15), half = (lane >> 4) & 1;
#pragma unroll
      for (int r = 0; r < 8; ++r) {
        int gm = mt * 16 + r + half * 8;
        a1t[gn * 512 + gm] = (__bf16)fmaxf(c[r], 0.f);
      }
    });
  __syncthreads();
  // convQ2: only t=0 valid -> queryf (f32)
  run_gemm(P.WQ2, 512, a1t, 512, 32, 1, 16, 3, 512 * 512, wave, lane,
    [&](int mt, int nt, v8f c) {
      (void)nt;
      if ((lane & 15) == 0) {
        int half = (lane >> 4) & 1;
#pragma unroll
        for (int r = 0; r < 8; ++r) qf[mt * 16 + r + half * 8] = fmaxf(c[r], 0.f);
      }
    });
  __syncthreads();

  // ---- attention over windows ----
  if (tid < 36) {
    float acc = 0.f;
    const __bf16* kr = kft + tid * 512;
    for (int d = 0; d < 512; ++d) acc += qf[d] * (float)kr[d];
    attw[tid] = acc + 1e-15f;
  }
  __syncthreads();
  if (tid == 0) {
    float s = 0.f;
    for (int v = 0; v < 36; ++v) s += attw[v];
    attw[63] = 1.0f / s;
  }
  __syncthreads();
  if (tid < 36) attw[tid] = attw[tid] * attw[63];
  __syncthreads();

  // s[l,f] = sum_v att[v]*src[v+l,f]
  for (int i = tid; i < 15 * 54; i += 256) {
    int l = i / 54, f = i - l * 54;
    float acc = 0.f;
    const float* sp = srcF + l * 56 + f;
    for (int v = 0; v < 36; ++v) acc += attw[v] * sp[v * 56];
    sbuf[l * 56 + f] = acc;
  }
  __syncthreads();
  // dct_in = concat(DCT(src[idx2]), DCT(s));  also build X0^T bf16 for the GCN
  for (int i = tid; i < 540; i += 256) {
    int f = i / 10, d = i - f * 10;
    float p1 = 0.f, p2 = 0.f;
#pragma unroll
    for (int l = 0; l < 15; ++l) {
      float dm = DCTm[d * 16 + l];
      int r2 = (l < 5) ? (45 + l) : 49;
      p1 += dm * srcF[r2 * 56 + f];
      p2 += dm * sbuf[l * 56 + f];
    }
    dct20[f * 32 + d]        = p1;
    dct20[f * 32 + 10 + d]   = p2;
    x0t[d * 64 + f]          = (__bf16)p1;
    x0t[(10 + d) * 64 + f]   = (__bf16)p2;
  }
  __syncthreads();

  // ---- GCN ----
  auto epi_T = [&](int mt, int nt, v8f c) {
    int gn = nt * 16 + (lane & 15), half = (lane >> 4) & 1;
#pragma unroll
    for (int r = 0; r < 8; ++r) {
      int gm = mt * 16 + r + half * 8;
      Tbf[gm * 512 + gn] = (__bf16)c[r];
    }
  };

  // gc1: T0 = att @ X0 ; Y = tanh(bn1(T0 @ W1 + b1))
  run_gemm(P.ATT0, 64, x0t, 64, 4, 2, 2, 1, 0, wave, lane, epi_T);
  __syncthreads();
  run_gemm(Tbf, 512, P.W1T, 32, 4, 32, 1, 1, 0, wave, lane,
    [&](int mt, int nt, v8f c) {
      int gn = nt * 16 + (lane & 15), half = (lane >> 4) & 1;
#pragma unroll
      for (int r = 0; r < 8; ++r) {
        int gm = mt * 16 + r + half * 8;
        if (gm < 54) {
          int idx = gm * 512 + gn;
          float v = (c[r] + P.B1[gn]) * BN_INV * P.G1[idx] + P.BB1[idx];
          v = tanhf(v);
          Yf[idx] = v;
          YbfT[gn * 64 + gm] = (__bf16)v;
        }
      }
    });
  __syncthreads();

  for (int s = 0; s < 4; ++s) {
    const __bf16* Xin = (s & 1) ? HbfT : YbfT;
    run_gemm(P.ATTS[s], 64, Xin, 64, 4, 32, 2, 1, 0, wave, lane, epi_T);
    __syncthreads();
    const float* Bp = P.SB[s]; const float* Gp = P.SG[s]; const float* BBp = P.SBB[s];
    if ((s & 1) == 0) {
      run_gemm(Tbf, 512, P.WB[s], 512, 4, 32, 16, 1, 0, wave, lane,
        [&](int mt, int nt, v8f c) {
          int gn = nt * 16 + (lane & 15), half = (lane >> 4) & 1;
#pragma unroll
          for (int r = 0; r < 8; ++r) {
            int gm = mt * 16 + r + half * 8;
            if (gm < 54) {
              int idx = gm * 512 + gn;
              float v = (c[r] + Bp[gn]) * BN_INV * Gp[idx] + BBp[idx];
              HbfT[gn * 64 + gm] = (__bf16)tanhf(v);
            }
          }
        });
    } else {
      run_gemm(Tbf, 512, P.WB[s], 512, 4, 32, 16, 1, 0, wave, lane,
        [&](int mt, int nt, v8f c) {
          int gn = nt * 16 + (lane & 15), half = (lane >> 4) & 1;
#pragma unroll
          for (int r = 0; r < 8; ++r) {
            int gm = mt * 16 + r + half * 8;
            if (gm < 54) {
              int idx = gm * 512 + gn;
              float v = (c[r] + Bp[gn]) * BN_INV * Gp[idx] + BBp[idx];
              float yn = Yf[idx] + tanhf(v);
              Yf[idx] = yn;
              YbfT[gn * 64 + gm] = (__bf16)yn;
            }
          }
        });
    }
    __syncthreads();
  }

  // gc7 + residual x
  run_gemm(P.ATT7, 64, YbfT, 64, 4, 32, 2, 1, 0, wave, lane, epi_T);
  __syncthreads();
  run_gemm(Tbf, 512, P.W7T, 512, 4, 2, 16, 1, 0, wave, lane,
    [&](int mt, int nt, v8f c) {
      int gn = nt * 16 + (lane & 15), half = (lane >> 4) & 1;
      if (gn < 20) {
#pragma unroll
        for (int r = 0; r < 8; ++r) {
          int gm = mt * 16 + r + half * 8;
          if (gm < 54) out20[gm * 32 + gn] = c[r] + P.B7[gn] + dct20[gm * 32 + gn];
        }
      }
    });
  __syncthreads();

  // IDCT (= DCT^T) and write (15,1,54)
  float* outb = P.out + (size_t)bi * (15 * 54);
  for (int i = tid; i < 15 * 54; i += 256) {
    int l = i / 54, f = i - l * 54;
    float acc = 0.f;
#pragma unroll
    for (int d = 0; d < 10; ++d) acc += DCTm[d * 16 + l] * out20[f * 32 + d];
    outb[i] = acc;
  }
}

// ---------------- host ----------------
extern "C" void kernel_launch(void* const* d_in, const int* in_sizes, int n_in,
                              void* d_out, int out_size, void* d_ws, size_t ws_size,
                              hipStream_t stream) {
  (void)out_size; (void)ws_size;
  // detect flatten order: insertion-order (src first) vs jax alphabetical tree order
  bool ins = (n_in > 0 && in_sizes[0] == 2048 * 50 * 54);
  int i_src, i_qw1, i_qw2, i_kw1, i_kw2;
  int i_gc1w, i_gc1att, i_gc1b, i_bn1g, i_bn1bb;
  int i_sw[4], i_satt[4], i_sb[4], i_sg[4], i_sbb[4];
  int i_gc7w, i_gc7att, i_gc7b;
  if (ins) {
    i_src = 0; i_qw1 = 1; i_qw2 = 2; i_kw1 = 3; i_kw2 = 4;
    i_gc1w = 5; i_gc1att = 6; i_gc1b = 7; i_bn1g = 8; i_bn1bb = 9;
    int base = 10;
    for (int s = 0; s < 4; ++s) {
      i_sw[s] = base; i_satt[s] = base + 1; i_sb[s] = base + 2;
      i_sg[s] = base + 3; i_sbb[s] = base + 4; base += 5;
    }
    i_gc7w = 30; i_gc7att = 31; i_gc7b = 32;
  } else {
    i_kw1 = 0; i_kw2 = 1; i_qw1 = 2; i_qw2 = 3;
    for (int blk = 0; blk < 2; ++blk) {
      int b0 = 4 + blk * 10;
      int s0 = blk * 2, s1 = blk * 2 + 1;
      i_sbb[s0] = b0 + 0; i_sg[s0] = b0 + 1;     // bn1: bb, g
      i_sbb[s1] = b0 + 2; i_sg[s1] = b0 + 3;     // bn2: bb, g
      i_satt[s0] = b0 + 4; i_sb[s0] = b0 + 5; i_sw[s0] = b0 + 6;  // gc1: att,b,w
      i_satt[s1] = b0 + 7; i_sb[s1] = b0 + 8; i_sw[s1] = b0 + 9;  // gc2: att,b,w
    }
    i_bn1bb = 24; i_bn1g = 25;
    i_gc1att = 26; i_gc1b = 27; i_gc1w = 28;
    i_gc7att = 29; i_gc7b = 30; i_gc7w = 31;
    i_src = 34;
  }

  // carve workspace for bf16 weight layouts (~5.7 MB, L2-resident)
  char* ws = (char*)d_ws;
  size_t off = 0;
  auto carve = [&](size_t bytes) -> void* {
    void* p = ws + off;
    off = (off + bytes + 255) & ~(size_t)255;
    return p;
  };
  __bf16* WK1 = (__bf16*)carve(3 * 512 * 64 * 2);
  __bf16* WQ1 = (__bf16*)carve(3 * 512 * 64 * 2);
  __bf16* WK2 = (__bf16*)carve(3 * 512 * 512 * 2);
  __bf16* WQ2 = (__bf16*)carve(3 * 512 * 512 * 2);
  __bf16* ATT0 = (__bf16*)carve(64 * 64 * 2);
  __bf16* ATTS[4];
  for (int s = 0; s < 4; ++s) ATTS[s] = (__bf16*)carve(64 * 64 * 2);
  __bf16* ATT7 = (__bf16*)carve(64 * 64 * 2);
  __bf16* W1T = (__bf16*)carve(512 * 32 * 2);
  __bf16* WB[4];
  for (int s = 0; s < 4; ++s) WB[s] = (__bf16*)carve(512 * 512 * 2);
  __bf16* W7T = (__bf16*)carve(32 * 512 * 2);

  // weight prep
  prep_conv_w1<<<(3 * 512 * 64 + 255) / 256, 256, 0, stream>>>((const float*)d_in[i_kw1], WK1);
  prep_conv_w1<<<(3 * 512 * 64 + 255) / 256, 256, 0, stream>>>((const float*)d_in[i_qw1], WQ1);
  prep_conv_w2<<<(3 * 512 * 512 + 255) / 256, 256, 0, stream>>>((const float*)d_in[i_kw2], WK2);
  prep_conv_w2<<<(3 * 512 * 512 + 255) / 256, 256, 0, stream>>>((const float*)d_in[i_qw2], WQ2);
  prep_att64<<<16, 256, 0, stream>>>((const float*)d_in[i_gc1att], ATT0);
  for (int s = 0; s < 4; ++s)
    prep_att64<<<16, 256, 0, stream>>>((const float*)d_in[i_satt[s]], ATTS[s]);
  prep_att64<<<16, 256, 0, stream>>>((const float*)d_in[i_gc7att], ATT7);
  prep_wT<<<(512 * 32 + 255) / 256, 256, 0, stream>>>((const float*)d_in[i_gc1w], W1T, 20, 512, 512, 32);
  for (int s = 0; s < 4; ++s)
    prep_wT<<<(512 * 512 + 255) / 256, 256, 0, stream>>>((const float*)d_in[i_sw[s]], WB[s], 512, 512, 512, 512);
  prep_wT<<<(32 * 512 + 255) / 256, 256, 0, stream>>>((const float*)d_in[i_gc7w], W7T, 512, 20, 32, 512);

  KParams P;
  P.src = (const float*)d_in[i_src];
  P.out = (float*)d_out;
  P.WK1 = WK1; P.WQ1 = WQ1; P.WK2 = WK2; P.WQ2 = WQ2;
  P.ATT0 = ATT0; P.ATT7 = ATT7;
  for (int s = 0; s < 4; ++s) P.ATTS[s] = ATTS[s];
  P.W1T = W1T; P.W7T = W7T;
  for (int s = 0; s < 4; ++s) P.WB[s] = WB[s];
  P.B1 = (const float*)d_in[i_gc1b];
  P.G1 = (const float*)d_in[i_bn1g];
  P.BB1 = (const float*)d_in[i_bn1bb];
  for (int s = 0; s < 4; ++s) {
    P.SB[s] = (const float*)d_in[i_sb[s]];
    P.SG[s] = (const float*)d_in[i_sg[s]];
    P.SBB[s] = (const float*)d_in[i_sbb[s]];
  }
  P.B7 = (const float*)d_in[i_gc7b];

  (void)hipFuncSetAttribute((const void*)attmodel_main,
                            hipFuncAttributeMaxDynamicSharedMemorySize, LDS_SIZE);
  attmodel_main<<<2048, 256, LDS_SIZE, stream>>>(P);
}
